// MultiHeadAttention_16853451669595
// MI455X (gfx1250) — matrix operand order repaired and checked by reading, not compiled, not run on hardware
//
#include <hip/hip_runtime.h>
#include <hip/hip_bf16.h>

#define DM   1024
#define HN   16
#define DKD  64
#define SEQ  2048
#define BATCH 4

typedef __attribute__((ext_vector_type(16))) __bf16 v16bf;
typedef __attribute__((ext_vector_type(8)))  __bf16 v8bf;
typedef __attribute__((ext_vector_type(8)))  float  v8f;
typedef __attribute__((ext_vector_type(4)))  unsigned int u32x4;
typedef __attribute__((ext_vector_type(8)))  int    i32x8;
typedef __attribute__((ext_vector_type(4)))  int    i32x4;

#if __has_builtin(__builtin_amdgcn_tensor_load_to_lds)
#define HAS_TDM 1
#else
#define HAS_TDM 0
#endif

#define LDS_STRIDE 80   // elems; 160B rows keep every 16B fragment chunk aligned

// ---- fragment helpers -------------------------------------------------------
// A-matrix 16x32 bf16 layout: lane<16 holds row M=lane, K={0..7,16..23};
// lane>=16 holds K={8..15,24..31}  ->  two 16B chunks at +half*8 and +16+half*8
__device__ __forceinline__ v16bf load_frag_A(const __bf16* rowp, int half) {
  union { v16bf v; v8bf h[2]; } u;
  u.h[0] = *(const v8bf*)(rowp + half * 8);
  u.h[1] = *(const v8bf*)(rowp + 16 + half * 8);
  return u.v;
}
// B-matrix 32x16 bf16 layout: lane<16 holds col N=lane, K={0..7,16..23}; lane>=16 K={8..15,24..31}
__device__ __forceinline__ v16bf load_frag_B(const __bf16* colp, int half) {
  union { v16bf v; v8bf h[2]; } u;
  u.h[0] = *(const v8bf*)(colp + half * 8);
  u.h[1] = *(const v8bf*)(colp + 16 + half * 8);
  return u.v;
}
__device__ __forceinline__ v8f wmma_bf16(v16bf a, v16bf b, v8f c) {
  return __builtin_amdgcn_wmma_f32_16x16x32_bf16(false, a, false, b, (short)0, c,
                                                 false, false);
}

// ---- Tensor Data Mover: 2D tile Global -> LDS (data_size = 2 bytes) --------
// 6-arg builtin form (amdgpu-toolchain / clang-23)
__device__ __forceinline__ void tdm_load_2d(unsigned lds_addr,
                                            unsigned long long gaddr,
                                            unsigned tdim0, unsigned tdim1,
                                            unsigned tile0, unsigned tile1,
                                            unsigned long long stride0) {
#if HAS_TDM
  u32x4 g0;
  g0[0] = 1u;                                            // count=1, user desc
  g0[1] = lds_addr;                                      // lds_addr [63:32]
  g0[2] = (unsigned)(gaddr & 0xffffffffull);             // global_addr lo
  g0[3] = (unsigned)((gaddr >> 32) & 0x1ffffffull)       // global_addr hi (56:32)
          | (2u << 30);                                  // type=2 ("image")
  i32x8 g1;
  g1[0] = (int)(1u << 16);                               // data_size=1 -> 2 bytes
  g1[1] = (int)(tdim0 << 16);                            // tensor_dim0 [79:48]
  g1[2] = (int)((tdim0 >> 16) | (tdim1 << 16));          // tensor_dim1 [111:80]
  g1[3] = (int)((tdim1 >> 16) | (tile0 << 16));          // tile_dim0 [127:112]
  g1[4] = (int)(tile1 & 0xffffu);                        // tile_dim1 [143:128]
  g1[5] = (int)(stride0 & 0xffffffffull);                // dim0_stride [207:160]
  g1[6] = (int)((stride0 >> 32) & 0xffffull);
  g1[7] = 0;
  i32x4 z4 = {0, 0, 0, 0};
  i32x8 z8 = {0, 0, 0, 0, 0, 0, 0, 0};
  __builtin_amdgcn_tensor_load_to_lds(g0, g1, z4, z4, z8, 0);
#else
  (void)lds_addr; (void)gaddr; (void)tdim0; (void)tdim1;
  (void)tile0; (void)tile1; (void)stride0;
#endif
}

// ---- shared GEMM tile staging (f32 -> bf16) --------------------------------
// 128x64 f32 tile, 256 threads: row = t>>1, 32 cols each (8 float4 loads)
__device__ __forceinline__ void stage_f32_tile(const float* __restrict__ src0,
                                               __bf16* dst, int t) {
  const int row = t >> 1, colb = (t & 1) * 32;
  const float4* src = (const float4*)(src0 + (size_t)row * DM + colb);
  __bf16* d0 = &dst[row * LDS_STRIDE + colb];
#pragma unroll
  for (int i = 0; i < 8; ++i) {
    float4 v = src[i];
    __bf16* d = d0 + i * 4;
    d[0] = (__bf16)v.x; d[1] = (__bf16)v.y;
    d[2] = (__bf16)v.z; d[3] = (__bf16)v.w;
  }
}

// ============================================================================
// Kernel 1: projection GEMM  out[m,n] = sum_k X[m,k]*W[n,k] + bias[n]
//   mode 0: store bf16 head-split [B,H,S,Dk]   (Q, K)
//   mode 1: store bf16 transposed [B,H,Dk,S]   (V)
// Block 256 threads (8 waves). Tile M=128, N=128, K-stage=64.
// Wave tile 32x64 -> 8 C frags, 16 WMMAs per staging round.
// ============================================================================
__global__ void __launch_bounds__(256)
proj_gemm_kernel(const float* __restrict__ X, const float* __restrict__ W,
                 const float* __restrict__ bias, __bf16* __restrict__ out,
                 int mode) {
  __shared__ __bf16 As[128 * LDS_STRIDE];   // 20KB
  __shared__ __bf16 Bs[128 * LDS_STRIDE];   // 20KB

  const int t = threadIdx.x;
  const int lane = t & 31, wave = t >> 5;
  const int wm = wave >> 1, wn = wave & 1;          // 4 x 2 wave grid
  const int half = lane >> 4, lr = lane & 15;
  const int m0 = blockIdx.y * 128, n0 = blockIdx.x * 128;

  v8f acc[2][4];
#pragma unroll
  for (int i = 0; i < 2; ++i)
#pragma unroll
    for (int j = 0; j < 4; ++j)
#pragma unroll
      for (int r = 0; r < 8; ++r) acc[i][j][r] = 0.0f;

  for (int kk = 0; kk < DM; kk += 64) {
    __syncthreads();
    __builtin_prefetch(X + (size_t)(m0 + (t >> 1)) * DM + kk + 64, 0, 0);
    stage_f32_tile(X + (size_t)m0 * DM + kk, As, t);
    stage_f32_tile(W + (size_t)n0 * DM + kk, Bs, t);
    __syncthreads();

#pragma unroll
    for (int ks = 0; ks < 2; ++ks) {
      v16bf af[2], bf[4];
#pragma unroll
      for (int tm = 0; tm < 2; ++tm)
        af[tm] = load_frag_A(&As[(wm * 32 + tm * 16 + lr) * LDS_STRIDE + ks * 32],
                             half);
#pragma unroll
      for (int tn = 0; tn < 4; ++tn)
        bf[tn] = load_frag_B(&Bs[(wn * 64 + tn * 16 + lr) * LDS_STRIDE + ks * 32],
                             half);
#pragma unroll
      for (int tm = 0; tm < 2; ++tm)
#pragma unroll
        for (int tn = 0; tn < 4; ++tn)
          acc[tm][tn] = wmma_bf16(af[tm], bf[tn], acc[tm][tn]);
    }
  }

  // epilogue: C layout -> (M = r + 8*half, N = lr)
#pragma unroll
  for (int tm = 0; tm < 2; ++tm)
#pragma unroll
    for (int tn = 0; tn < 4; ++tn) {
      const int colg = n0 + wn * 64 + tn * 16 + lr;
      const float bv = bias[colg];
#pragma unroll
      for (int r = 0; r < 8; ++r) {
        const int rowg = m0 + wm * 32 + tm * 16 + r + 8 * half;
        const float val = acc[tm][tn][r] + bv;
        const int b = rowg >> 11, s = rowg & (SEQ - 1);
        const int h = colg >> 6, dk = colg & (DKD - 1);
        size_t idx;
        if (mode == 0)
          idx = (((size_t)(b * HN + h) * SEQ) + s) * DKD + dk;   // [B,H,S,Dk]
        else
          idx = (((size_t)(b * HN + h) * DKD) + dk) * SEQ + s;   // [B,H,Dk,S]
        out[idx] = (__bf16)val;
      }
    }
}

// ============================================================================
// Kernel 2: flash attention. Block = 128 threads (4 waves), each wave owns a
// 16-query tile; iterate over 32-key tiles. K/V tiles staged via TDM.
// Qh,Kh: [B,H,S,Dk] bf16; Vt: [B,H,Dk,S] bf16; Ab out: [B,S,H*Dk] bf16.
// ============================================================================
__global__ void __launch_bounds__(128)
flash_attn_kernel(const __bf16* __restrict__ Qh, const __bf16* __restrict__ Kh,
                  const __bf16* __restrict__ Vt, __bf16* __restrict__ Ab) {
  __shared__ __bf16 Kt[32 * DKD];        // [key][dk]   4KB
  __shared__ __bf16 Vl[DKD * 32];        // [dk][key]   4KB
  __shared__ __bf16 Pl[4 * 16 * 32];     // per-wave P scratch 4KB

  const int t = threadIdx.x;
  const int lane = t & 31, wave = t >> 5;
  const int half = lane >> 4, lr = lane & 15;
  const int bh = blockIdx.y;
  const int b = bh >> 4, h = bh & (HN - 1);
  const size_t base = (size_t)bh * SEQ * DKD;
  const __bf16* Qb = Qh + base;
  const __bf16* Kb = Kh + base;
  const __bf16* Vb = Vt + base;          // [Dk][S]
  const int q0 = blockIdx.x * 64 + wave * 16;

  // Q fragments (Dk = 64 -> two 16x32 A-frags), loaded once from global
  v16bf qf[2];
#pragma unroll
  for (int j = 0; j < 2; ++j)
    qf[j] = load_frag_A(Qb + (size_t)(q0 + lr) * DKD + j * 32, half);

  float m_r[8], l_r[8];
  v8f accO[4];
#pragma unroll
  for (int r = 0; r < 8; ++r) { m_r[r] = -3.0e38f; l_r[r] = 0.0f; }
#pragma unroll
  for (int d = 0; d < 4; ++d)
#pragma unroll
    for (int r = 0; r < 8; ++r) accO[d][r] = 0.0f;

  for (int kt = 0; kt < SEQ; kt += 32) {
#if HAS_TDM
    if (wave == 0) {
      const unsigned klds = (unsigned)(size_t)&Kt[0];
      const unsigned vlds = (unsigned)(size_t)&Vl[0];
      // K tile: 32 rows x 64 dk, row stride 64
      tdm_load_2d(klds, (unsigned long long)(size_t)(Kb + (size_t)kt * DKD),
                  DKD, SEQ, DKD, 32, DKD);
      // V tile: 64 dk rows x 32 keys from Vt, row stride SEQ
      tdm_load_2d(vlds, (unsigned long long)(size_t)(Vb + kt),
                  SEQ, DKD, 32, DKD, SEQ);
      __builtin_amdgcn_s_wait_tensorcnt(0);
    }
#else
    {  // cooperative fallback: contiguous K copy + strided V rows
      const uint4* ks = (const uint4*)(Kb + (size_t)kt * DKD);
      uint4* kd = (uint4*)&Kt[0];
#pragma unroll
      for (int i = 0; i < 2; ++i) kd[t * 2 + i] = ks[t * 2 + i];
      const int row = t >> 1, chunk = (t & 1);
      const uint4* vs =
          (const uint4*)(Vb + (size_t)row * SEQ + kt + chunk * 16);
      uint4* vd = (uint4*)&Vl[row * 32 + chunk * 16];
#pragma unroll
      for (int i = 0; i < 2; ++i) vd[i] = vs[i];
    }
#endif
    __syncthreads();

    // scores 16x32 = two 16x16 C frags; contraction over Dk=64 (2 WMMA each)
    v8f sc[2];
#pragma unroll
    for (int nt = 0; nt < 2; ++nt) {
#pragma unroll
      for (int r = 0; r < 8; ++r) sc[nt][r] = 0.0f;
#pragma unroll
      for (int j = 0; j < 2; ++j) {
        v16bf kfrag = load_frag_B(&Kt[(nt * 16 + lr) * DKD + j * 32], half);
        sc[nt] = wmma_bf16(qf[j], kfrag, sc[nt]);
      }
#pragma unroll
      for (int r = 0; r < 8; ++r) sc[nt][r] *= 0.125f;  // 1/sqrt(64)
    }

    // online softmax per row (row = r + 8*half, cols across 16-lane group)
#pragma unroll
    for (int r = 0; r < 8; ++r) {
      float tmax = fmaxf(sc[0][r], sc[1][r]);
#pragma unroll
      for (int m = 1; m < 16; m <<= 1)
        tmax = fmaxf(tmax, __shfl_xor(tmax, m, 32));
      const float mnew = fmaxf(m_r[r], tmax);
      const float sold = __expf(m_r[r] - mnew);
      const float p0 = __expf(sc[0][r] - mnew);
      const float p1 = __expf(sc[1][r] - mnew);
      float rs = p0 + p1;
#pragma unroll
      for (int m = 1; m < 16; m <<= 1) rs += __shfl_xor(rs, m, 32);
      l_r[r] = l_r[r] * sold + rs;
      m_r[r] = mnew;
      sc[0][r] = p0; sc[1][r] = p1;
#pragma unroll
      for (int d = 0; d < 4; ++d) accO[d][r] *= sold;
    }

    // P: C layout -> A layout via per-wave LDS scratch
    __bf16* pw = &Pl[wave * 16 * 32];
#pragma unroll
    for (int nt = 0; nt < 2; ++nt)
#pragma unroll
      for (int r = 0; r < 8; ++r)
        pw[(r + 8 * half) * 32 + nt * 16 + lr] = (__bf16)sc[nt][r];
    asm volatile("s_wait_dscnt 0" ::: "memory");
    v16bf pf = load_frag_A(&pw[lr * 32], half);

    // O += P (16x32) @ V (32x64): 4 N-tiles of 16
#pragma unroll
    for (int d = 0; d < 4; ++d) {
      v16bf vfrag = load_frag_B(&Vl[(d * 16 + lr) * 32], half);
      accO[d] = wmma_bf16(pf, vfrag, accO[d]);
    }
    __syncthreads();
  }

  // epilogue: normalize and store to [B,S,H*Dk]
#pragma unroll
  for (int r = 0; r < 8; ++r) {
    const float invl = 1.0f / l_r[r];
    const int srow = q0 + r + 8 * half;
#pragma unroll
    for (int d = 0; d < 4; ++d)
      Ab[((size_t)(b * SEQ + srow)) * DM + h * DKD + d * 16 + lr] =
          (__bf16)(accO[d][r] * invl);
  }
}

// ============================================================================
// Kernel 3: output projection  d_out[m,n] = sum_k Ab[m,k]*Wo[n,k] + b_o[n]
// A already bf16; W converted on the fly; f32 output. Same tiling as kernel 1.
// ============================================================================
__global__ void __launch_bounds__(256)
out_gemm_kernel(const __bf16* __restrict__ A, const float* __restrict__ W,
                const float* __restrict__ bias, float* __restrict__ out) {
  __shared__ __bf16 As[128 * LDS_STRIDE];
  __shared__ __bf16 Bs[128 * LDS_STRIDE];

  const int t = threadIdx.x;
  const int lane = t & 31, wave = t >> 5;
  const int wm = wave >> 1, wn = wave & 1;
  const int half = lane >> 4, lr = lane & 15;
  const int m0 = blockIdx.y * 128, n0 = blockIdx.x * 128;

  v8f acc[2][4];
#pragma unroll
  for (int i = 0; i < 2; ++i)
#pragma unroll
    for (int j = 0; j < 4; ++j)
#pragma unroll
      for (int r = 0; r < 8; ++r) acc[i][j][r] = 0.0f;

  for (int kk = 0; kk < DM; kk += 64) {
    __syncthreads();
    {  // stage A tile 128x64 bf16 directly (32 elems = 4 v8bf per thread)
      const int row = t >> 1, colb = (t & 1) * 32;
      const v8bf* src = (const v8bf*)(A + (size_t)(m0 + row) * DM + kk + colb);
      v8bf* dst = (v8bf*)&As[row * LDS_STRIDE + colb];
#pragma unroll
      for (int i = 0; i < 4; ++i) dst[i] = src[i];
    }
    stage_f32_tile(W + (size_t)n0 * DM + kk, Bs, t);
    __syncthreads();

#pragma unroll
    for (int ks = 0; ks < 2; ++ks) {
      v16bf af[2], bf[4];
#pragma unroll
      for (int tm = 0; tm < 2; ++tm)
        af[tm] = load_frag_A(&As[(wm * 32 + tm * 16 + lr) * LDS_STRIDE + ks * 32],
                             half);
#pragma unroll
      for (int tn = 0; tn < 4; ++tn)
        bf[tn] = load_frag_B(&Bs[(wn * 64 + tn * 16 + lr) * LDS_STRIDE + ks * 32],
                             half);
#pragma unroll
      for (int tm = 0; tm < 2; ++tm)
#pragma unroll
        for (int tn = 0; tn < 4; ++tn)
          acc[tm][tn] = wmma_bf16(af[tm], bf[tn], acc[tm][tn]);
    }
  }

#pragma unroll
  for (int tm = 0; tm < 2; ++tm)
#pragma unroll
    for (int tn = 0; tn < 4; ++tn) {
      const int colg = n0 + wn * 64 + tn * 16 + lr;
      const float bv = bias[colg];
#pragma unroll
      for (int r = 0; r < 8; ++r) {
        const int rowg = m0 + wm * 32 + tm * 16 + r + 8 * half;
        out[(size_t)rowg * DM + colg] = acc[tm][tn][r] + bv;
      }
    }
}

// ============================================================================
extern "C" void kernel_launch(void* const* d_in, const int* in_sizes, int n_in,
                              void* d_out, int out_size, void* d_ws,
                              size_t ws_size, hipStream_t stream) {
  const float* q   = (const float*)d_in[0];
  const float* k   = (const float*)d_in[1];
  const float* v   = (const float*)d_in[2];
  const float* w_q = (const float*)d_in[3];
  const float* b_q = (const float*)d_in[4];
  const float* w_k = (const float*)d_in[5];
  const float* b_k = (const float*)d_in[6];
  const float* w_v = (const float*)d_in[7];
  const float* b_v = (const float*)d_in[8];
  const float* w_o = (const float*)d_in[9];
  const float* b_o = (const float*)d_in[10];
  float* out = (float*)d_out;

  const size_t NE = (size_t)BATCH * SEQ * DM;  // 8.4M elems, 16MB as bf16
  __bf16* Qh = (__bf16*)d_ws;
  __bf16* Kh = Qh + NE;
  __bf16* Vt = Kh + NE;
  __bf16* Ab = Vt + NE;

  const dim3 gemm_grid(DM / 128, (BATCH * SEQ) / 128);  // (8, 64)
  proj_gemm_kernel<<<gemm_grid, 256, 0, stream>>>(q, w_q, b_q, Qh, 0);
  proj_gemm_kernel<<<gemm_grid, 256, 0, stream>>>(k, w_k, b_k, Kh, 0);
  proj_gemm_kernel<<<gemm_grid, 256, 0, stream>>>(v, w_v, b_v, Vt, 1);

  const dim3 fa_grid(SEQ / 64, BATCH * HN);             // (32, 64)
  flash_attn_kernel<<<fa_grid, 128, 0, stream>>>(Qh, Kh, Vt, Ab);

  out_gemm_kernel<<<gemm_grid, 256, 0, stream>>>(Ab, w_o, b_o, out);
}
